// LSTM11__43164421324811
// MI455X (gfx1250) — matrix operand order repaired and checked by reading, not compiled, not run on hardware
//
#include <hip/hip_runtime.h>
#include <hip/hip_bf16.h>

// CDNA5 / gfx1250: wave32, WMMA 16x16x32 f16 -> f32.
//
// Strategy: weights live in the A operand (constant over all 2048 steps,
// VGPR-resident); per-step state [x_t;h0] / [h0;h1] is packed into the B
// operand (K x 16, columns = 16 batch elements). Gate groups i/f/g/o are
// padded to 16 rows each so activations are clean elementwise ops on the
// f32 C tiles. Bias is preloaded into C. One wave = 16 batch elements,
// 256 waves total, 8 WMMAs + ~activations per timestep.

typedef __attribute__((ext_vector_type(16))) _Float16 v16h;
typedef __attribute__((ext_vector_type(8)))  float    v8f;

#define T_LEN  2048
#define B_TOT  4096
#define HID    10
#define TAIL0  608      // T - 1440 : first timestep that feeds the output
#define OUT_T  1439     // number of output timesteps per batch row

__device__ __forceinline__ float fsig(float x) {
    // sigmoid via v_exp_f32 + v_rcp_f32
    float e = __expf(-x);
    return __builtin_amdgcn_rcpf(1.0f + e);
}
__device__ __forceinline__ float ftanh_(float x) {
    // tanh(x) = 2*sigmoid(2x) - 1
    return __builtin_fmaf(2.0f, fsig(2.0f * x), -1.0f);
}

#define WMMA_F16(A, B, C) \
    __builtin_amdgcn_wmma_f32_16x16x32_f16(false, (A), false, (B), (short)0, (C), false, false)

__global__ __launch_bounds__(32, 1) void lstm2_wmma_fused(
    const float* __restrict__ x,
    const float* __restrict__ w_ih0, const float* __restrict__ w_hh0,
    const float* __restrict__ b_ih0, const float* __restrict__ b_hh0,
    const float* __restrict__ w_ih1, const float* __restrict__ w_hh1,
    const float* __restrict__ b_ih1, const float* __restrict__ b_hh1,
    const float* __restrict__ fc1_w, const float* __restrict__ fc1_b,
    const float* __restrict__ fc2_w, const float* __restrict__ fc2_b,
    float* __restrict__ out)
{
    const int  lane = threadIdx.x & 31;
    const int  col  = lane & 15;      // batch column (B/C/D) == row M (A operand)
    const bool hi   = lane >= 16;
    const int  bidx = blockIdx.x * 16 + col;

    // ---------------- A operands: 4 gate tiles per layer (16x32 f16) ----------
    // A element e of lane maps to K = e + 16*hi; row M = col.
    // Layer0 K: 0 -> w_ih0 (input dim 1), 1..10 -> w_hh0, rest 0.
    // Layer1 K: 0..9 -> w_ih1, 10..19 -> w_hh1, rest 0.
    v16h A0[4], A1[4];
    v8f  Cb0[4], Cb1[4];
    #pragma unroll
    for (int g = 0; g < 4; ++g) {
        #pragma unroll
        for (int e = 0; e < 16; ++e) {
            const int k = e + (hi ? 16 : 0);
            float w0 = 0.0f, w1 = 0.0f;
            if (col < HID) {
                const int j = g * HID + col;           // PyTorch gate order i,f,g,o
                if (k == 0)            w0 = w_ih0[j];
                else if (k <= HID)     w0 = w_hh0[j * HID + (k - 1)];
                if (k < HID)           w1 = w_ih1[j * HID + k];
                else if (k < 2 * HID)  w1 = w_hh1[j * HID + (k - HID)];
            }
            A0[g][e] = (_Float16)w0;
            A1[g][e] = (_Float16)w1;
        }
        // Bias preloaded into the C operand (C row = r + 8*hi).
        #pragma unroll
        for (int r = 0; r < 8; ++r) {
            const int row = r + (hi ? 8 : 0);
            float bb0 = 0.0f, bb1 = 0.0f;
            if (row < HID) {
                bb0 = b_ih0[g * HID + row] + b_hh0[g * HID + row];
                bb1 = b_ih1[g * HID + row] + b_hh1[g * HID + row];
            }
            Cb0[g][r] = bb0;
            Cb1[g][r] = bb1;
        }
    }

    // ---------------- FC weights (register resident, split by lane half) ------
    // lo lanes reduce h1 rows 0..7; hi lanes reduce rows 8..9.
    float fw[6][8], f1b[6], f2w[6];
    #pragma unroll
    for (int kk = 0; kk < 6; ++kk) {
        #pragma unroll
        for (int j = 0; j < 8; ++j) {
            float v = 0.0f;
            if (hi) { if (j < 2) v = fc1_w[kk * HID + 8 + j]; }
            else               v = fc1_w[kk * HID + j];
            fw[kk][j] = v;
        }
        f1b[kk] = fc1_b[kk];
        f2w[kk] = fc2_w[kk];
    }
    const float f2b = fc2_b[0];

    // ---------------- Recurrent state in C-layout (rows=hidden, cols=batch) ---
    float h0r[8], c0r[8], h1r[8], c1r[8];
    #pragma unroll
    for (int r = 0; r < 8; ++r) { h0r[r] = 0.f; c0r[r] = 0.f; h1r[r] = 0.f; c1r[r] = 0.f; }

    const float* xrow = x   + (size_t)bidx * T_LEN;
    float*       orow = out + (size_t)bidx * OUT_T;

    #pragma unroll 1
    for (int t0 = 0; t0 < T_LEN; t0 += 4) {
        const float4 xq = *(const float4*)(xrow + t0);
        if (t0 + 64 < T_LEN) __builtin_prefetch(xrow + t0 + 64, 0, 0);  // global_prefetch

        #pragma unroll
        for (int s = 0; s < 4; ++s) {
            const int   t  = t0 + s;
            const float xv = (s == 0) ? xq.x : (s == 1) ? xq.y : (s == 2) ? xq.z : xq.w;

            // ======== layer 0: B = [x_t ; h0] (K=0..10), hi-lane K-range all zero
            const float h0_8 = __shfl_xor(h0r[0], 16, 32);   // rows 8,9 live in other half
            const float h0_9 = __shfl_xor(h0r[1], 16, 32);
            v16h B0;
            #pragma unroll
            for (int e = 0; e < 16; ++e) B0[e] = (_Float16)0.f;
            if (!hi) {
                B0[0] = (_Float16)xv;
                #pragma unroll
                for (int j = 0; j < 8; ++j) B0[1 + j] = (_Float16)h0r[j];
                B0[9]  = (_Float16)h0_8;
                B0[10] = (_Float16)h0_9;
            }
            v8f Gi = WMMA_F16(A0[0], B0, Cb0[0]);
            v8f Gf = WMMA_F16(A0[1], B0, Cb0[1]);
            v8f Gg = WMMA_F16(A0[2], B0, Cb0[2]);
            v8f Go = WMMA_F16(A0[3], B0, Cb0[3]);
            #pragma unroll
            for (int r = 0; r < 8; ++r) {
                const float ig = fsig(Gi[r]);
                const float fg = fsig(Gf[r]);
                const float gg = ftanh_(Gg[r]);
                const float og = fsig(Go[r]);
                const float c  = __builtin_fmaf(fg, c0r[r], ig * gg);
                c0r[r] = c;
                h0r[r] = og * ftanh_(c);
            }

            // ======== layer 1: B = [h0_new (K=0..9) ; h1_prev (K=10..19)]
            const float h0n8 = __shfl_xor(h0r[0], 16, 32);
            const float h0n9 = __shfl_xor(h0r[1], 16, 32);
            const float h1s6 = __shfl_xor(h1r[6], 16, 32);   // hi lanes fetch rows 6,7
            const float h1s7 = __shfl_xor(h1r[7], 16, 32);
            v16h B1;
            #pragma unroll
            for (int e = 0; e < 16; ++e) B1[e] = (_Float16)0.f;
            if (!hi) {
                #pragma unroll
                for (int j = 0; j < 8; ++j) B1[j] = (_Float16)h0r[j];
                B1[8] = (_Float16)h0n8;
                B1[9] = (_Float16)h0n9;
                #pragma unroll
                for (int j = 0; j < 6; ++j) B1[10 + j] = (_Float16)h1r[j];
            } else {
                B1[0] = (_Float16)h1s6;     // K=16 -> h1 row 6
                B1[1] = (_Float16)h1s7;     // K=17 -> h1 row 7
                B1[2] = (_Float16)h1r[0];   // K=18 -> h1 row 8
                B1[3] = (_Float16)h1r[1];   // K=19 -> h1 row 9
            }
            Gi = WMMA_F16(A1[0], B1, Cb1[0]);
            Gf = WMMA_F16(A1[1], B1, Cb1[1]);
            Gg = WMMA_F16(A1[2], B1, Cb1[2]);
            Go = WMMA_F16(A1[3], B1, Cb1[3]);
            #pragma unroll
            for (int r = 0; r < 8; ++r) {
                const float ig = fsig(Gi[r]);
                const float fg = fsig(Gf[r]);
                const float gg = ftanh_(Gg[r]);
                const float og = fsig(Go[r]);
                const float c  = __builtin_fmaf(fg, c1r[r], ig * gg);
                c1r[r] = c;
                h1r[r] = og * ftanh_(c);
            }

            // ======== fused FC epilogue for t in [608, 2046]
            if (t >= TAIL0 && t < T_LEN - 1) {
                float acc[6];
                #pragma unroll
                for (int kk = 0; kk < 6; ++kk) {
                    float a = 0.0f;
                    #pragma unroll
                    for (int j = 0; j < 8; ++j) a = __builtin_fmaf(h1r[j], fw[kk][j], a);
                    acc[kk] = a;
                }
                #pragma unroll
                for (int kk = 0; kk < 6; ++kk)
                    acc[kk] += __shfl_xor(acc[kk], 16, 32);   // both halves get full sum
                if (!hi) {
                    float o = f2b;
                    #pragma unroll
                    for (int kk = 0; kk < 6; ++kk)
                        o = __builtin_fmaf(ftanh_(acc[kk] + f1b[kk]), f2w[kk], o);
                    orow[t - TAIL0] = o;
                }
            }
        }
    }
}

extern "C" void kernel_launch(void* const* d_in, const int* in_sizes, int n_in,
                              void* d_out, int out_size, void* d_ws, size_t ws_size,
                              hipStream_t stream) {
    (void)in_sizes; (void)n_in; (void)d_ws; (void)ws_size; (void)out_size;
    const float* x     = (const float*)d_in[0];
    const float* w_ih0 = (const float*)d_in[1];
    const float* w_hh0 = (const float*)d_in[2];
    const float* b_ih0 = (const float*)d_in[3];
    const float* b_hh0 = (const float*)d_in[4];
    const float* w_ih1 = (const float*)d_in[5];
    const float* w_hh1 = (const float*)d_in[6];
    const float* b_ih1 = (const float*)d_in[7];
    const float* b_hh1 = (const float*)d_in[8];
    const float* fc1_w = (const float*)d_in[9];
    const float* fc1_b = (const float*)d_in[10];
    const float* fc2_w = (const float*)d_in[11];
    const float* fc2_b = (const float*)d_in[12];
    float* out = (float*)d_out;

    dim3 grid(B_TOT / 16);   // 256 independent single-wave workgroups
    dim3 block(32);          // one wave32 per workgroup
    lstm2_wmma_fused<<<grid, block, 0, stream>>>(
        x, w_ih0, w_hh0, b_ih0, b_hh0,
        w_ih1, w_hh1, b_ih1, b_hh1,
        fc1_w, fc1_b, fc2_w, fc2_b, out);
}
